// Head_61426622267718
// MI455X (gfx1250) — compile-verified, matrix operand order checked by reading, not run on hardware
//
#include <hip/hip_runtime.h>

typedef __attribute__((ext_vector_type(16))) _Float16 v16h;
typedef __attribute__((ext_vector_type(8)))  _Float16 v8h;
typedef __attribute__((ext_vector_type(8)))  float    v8f;

constexpr int B = 4, S = 4096, E = 2048, H = 128;
constexpr float SCALE = 0.08838834764831845f;  // 128^-0.5

__device__ __forceinline__ v8f wmma16(v16h a, v16h b, v8f c) {
  // v_wmma_f32_16x16x32_f16: D = A(16x32 f16) * B(32x16 f16) + C(16x16 f32)
  return __builtin_amdgcn_wmma_f32_16x16x32_f16(false, a, false, b, (short)0, c,
                                                false, false);
}

// Async copy 16B/lane from global to LDS (ASYNCcnt-tracked, GV addressing).
__device__ __forceinline__ void async_ld_b128(unsigned lds_off, const void* g) {
  asm volatile("global_load_async_to_lds_b128 %0, %1, off"
               :: "v"(lds_off), "v"(g) : "memory");
}

// A-operand (and B-operand via B^T) fragment: 16x32 f16 tile, row-major source.
// Per ISA: lane m=L&15, half=L>>4; elements [0..7]=row m cols half*8+0..7,
// elements [8..15]=row m cols 16+half*8+0..7. Works for global or LDS pointers.
__device__ __forceinline__ v16h frag_f16(const _Float16* p, int ld, int lane) {
  const int m = lane & 15, hf = lane >> 4;
  const _Float16* r = p + (size_t)m * ld + hf * 8;
  v8h lo = *(const v8h*)(r);
  v8h hi = *(const v8h*)(r + 16);
  v16h o;
#pragma unroll
  for (int i = 0; i < 8; ++i) { o[i] = lo[i]; o[8 + i] = hi[i]; }
  return o;
}

// Same fragment but from fp32 source, converting to f16 in registers.
__device__ __forceinline__ v16h frag_f32(const float* p, int ld, int lane) {
  const int m = lane & 15, hf = lane >> 4;
  const float* r = p + (size_t)m * ld + hf * 8;
  v16h o;
#pragma unroll
  for (int i = 0; i < 8; ++i) {
    o[i]     = (_Float16)r[i];
    o[8 + i] = (_Float16)r[16 + i];
  }
  return o;
}

// W[E][H] fp32 -> WT[H][E] f16
__global__ void wtrans_kernel(const float* __restrict__ W, _Float16* __restrict__ WT) {
  int i = blockIdx.x * 256 + threadIdx.x;
  if (i >= E * H) return;
  int h = i / E, e = i % E;
  WT[i] = (_Float16)W[(size_t)e * H + h];
}

// Y[M,H] = X[M,E] @ WT^T ; one wave computes a 16x128 output strip.
// storeT=1 writes Y transposed per-batch as vhT[b][h][s].
__global__ __launch_bounds__(128) void proj_kernel(const float* __restrict__ X,
                                                   const _Float16* __restrict__ WT,
                                                   _Float16* __restrict__ Y,
                                                   int storeT) {
  const int lane = threadIdx.x & 31;
  const int wave = threadIdx.x >> 5;
  const int row0 = (blockIdx.x * 4 + wave) * 16;

  v8f acc[8] = {};
  for (int k0 = 0; k0 < E; k0 += 32) {
    v16h a = frag_f32(X + (size_t)row0 * E + k0, E, lane);
#pragma unroll
    for (int nt = 0; nt < 8; ++nt) {
      v16h b = frag_f16(WT + (size_t)(nt * 16) * E + k0, E, lane);
      acc[nt] = wmma16(a, b, acc[nt]);
    }
  }
  // C layout: VGPR r holds row m=r+8*half, col n=lane&15
  const int n = lane & 15, hf = lane >> 4;
  if (!storeT) {
#pragma unroll
    for (int nt = 0; nt < 8; ++nt)
#pragma unroll
      for (int r = 0; r < 8; ++r) {
        const int m = r + 8 * hf;
        Y[(size_t)(row0 + m) * H + nt * 16 + n] = (_Float16)acc[nt][r];
      }
  } else {
    const int b_ = row0 / S, s0 = row0 % S;
#pragma unroll
    for (int nt = 0; nt < 8; ++nt)
#pragma unroll
      for (int r = 0; r < 8; ++r) {
        const int m = r + 8 * hf;
        Y[(size_t)b_ * H * S + (size_t)(nt * 16 + n) * S + (s0 + m)] =
            (_Float16)acc[nt][r];
      }
  }
}

// Flash attention: grid (S/64, B), 128 threads; each wave owns 16 query rows.
// K strips (32x128) and V^T strips (128x32) are double-buffered in LDS via
// GLOBAL_LOAD_ASYNC_TO_LDS_B128: strip t+32 streams in while t is consumed.
__global__ __launch_bounds__(128) void attn_kernel(const _Float16* __restrict__ qh,
                                                   const _Float16* __restrict__ kh,
                                                   const _Float16* __restrict__ vhT,
                                                   float* __restrict__ out) {
  __shared__ _Float16 kTile[2][32][128];  // K strips, row-major [t][h]   (16 KB)
  __shared__ _Float16 vTile[2][128][32];  // V^T strips, row-major [h][t] (16 KB)
  __shared__ _Float16 pTile[4][16 * 32];  // per-wave P re-layout buffers  (4 KB)

  const int tid  = threadIdx.x;
  const int lane = tid & 31;
  const int wave = tid >> 5;
  const int b    = blockIdx.y;
  const int q0   = blockIdx.x * 64 + wave * 16;

  const _Float16* qB = qh + (size_t)b * S * H;
  const _Float16* kB = kh + (size_t)b * S * H;
  const _Float16* vB = vhT + (size_t)b * H * S;

  unsigned kBase[2], vBase[2];
  kBase[0] = (unsigned)(size_t)(&kTile[0][0][0]);
  kBase[1] = (unsigned)(size_t)(&kTile[1][0][0]);
  vBase[0] = (unsigned)(size_t)(&vTile[0][0][0]);
  vBase[1] = (unsigned)(size_t)(&vTile[1][0][0]);

  // issue one strip: 8 async b128 ops per thread (K 8KB contiguous + V^T rows)
  auto stage_strip = [&](int t, int buf) {
#pragma unroll
    for (int i = 0; i < 4; ++i) {
      const int c = tid + 128 * i;  // 16-byte chunk id 0..511
      async_ld_b128(kBase[buf] + c * 16, kB + (size_t)t * H + c * 8);
    }
#pragma unroll
    for (int i = 0; i < 4; ++i) {
      const int c = tid + 128 * i;
      const int row = c >> 2, sub = c & 3;
      async_ld_b128(vBase[buf] + c * 16, vB + (size_t)row * S + t + sub * 8);
    }
  };

  v16h qf[4];
#pragma unroll
  for (int kk = 0; kk < 4; ++kk)
    qf[kk] = frag_f16(qB + (size_t)q0 * H + kk * 32, H, lane);

  v8f o[8] = {};
  float mrow[8], lrow[8];
#pragma unroll
  for (int r = 0; r < 8; ++r) { mrow[r] = -__builtin_inff(); lrow[r] = 0.0f; }

  const int n = lane & 15, hf = lane >> 4;
  // Block-uniform causal bound: last strip touching row q0_blk+63.
  const int tLastBlk = ((blockIdx.x * 64 + 63) >> 5) << 5;

  stage_strip(0, 0);  // prologue: strip 0 -> buffer 0

  for (int t = 0; t <= tLastBlk; t += 32) {
    const int buf = (t >> 5) & 1;
    const bool hasNext = (t + 32 <= tLastBlk);  // block-uniform

    __syncthreads();  // readers of buf^1 (iter t-32) are done; safe to refill
    if (hasNext) {
      stage_strip(t + 32, buf ^ 1);
      // async loads complete in order: <=8 outstanding => current strip landed
      asm volatile("s_wait_asynccnt 0x8" ::: "memory");
    } else {
      asm volatile("s_wait_asynccnt 0x0" ::: "memory");
    }
    __syncthreads();  // all waves' strip-t data visible in LDS

    if (t <= q0 + 15) {  // wave-uniform: strip intersects this wave's rows
      // ---- scores: S[16 x 32] = Q(16xH) . K^T, as two 16x16 C tiles ----
      v8f s0 = {}, s1 = {};
#pragma unroll
      for (int kk = 0; kk < 4; ++kk) {
        v16h k0f = frag_f16(&kTile[buf][0][kk * 32], 128, lane);
        v16h k1f = frag_f16(&kTile[buf][16][kk * 32], 128, lane);
        s0 = wmma16(qf[kk], k0f, s0);
        s1 = wmma16(qf[kk], k1f, s1);
      }

      // ---- online softmax (rows live in 16-lane half-groups) ----
      const bool needMask = (t + 31 > q0);
      float scl[8];
#pragma unroll
      for (int r = 0; r < 8; ++r) {
        const int row = q0 + r + 8 * hf;
        float a = s0[r] * SCALE;
        float c = s1[r] * SCALE;
        if (needMask) {
          if (t + n > row)      a = -__builtin_inff();
          if (t + 16 + n > row) c = -__builtin_inff();
        }
        float vmx = fmaxf(a, c);
#pragma unroll
        for (int off = 1; off < 16; off <<= 1)
          vmx = fmaxf(vmx, __shfl_xor(vmx, off, 32));
        const float mNew = fmaxf(mrow[r], vmx);
        const float p0 = __expf(a - mNew);
        const float p1 = __expf(c - mNew);
        float ps = p0 + p1;
#pragma unroll
        for (int off = 1; off < 16; off <<= 1) ps += __shfl_xor(ps, off, 32);
        scl[r]  = __expf(mrow[r] - mNew);
        lrow[r] = lrow[r] * scl[r] + ps;
        mrow[r] = mNew;
        const int m = r + 8 * hf;
        pTile[wave][m * 32 + n]      = (_Float16)p0;
        pTile[wave][m * 32 + 16 + n] = (_Float16)p1;
      }

      // C-layout -> A-layout re-shape of P through LDS (same wave, DS in-order)
      asm volatile("s_wait_dscnt 0" ::: "memory");
      v16h pf;
      {
        const int m = lane & 15;
        const _Float16* rp = &pTile[wave][m * 32 + hf * 8];
        v8h lo = *(const v8h*)(rp);
        v8h hi = *(const v8h*)(rp + 16);
#pragma unroll
        for (int i = 0; i < 8; ++i) { pf[i] = lo[i]; pf[8 + i] = hi[i]; }
      }

      // ---- O = O*scale + P(16x32) . V(32x128); V^T rows from LDS ----
#pragma unroll
      for (int ct = 0; ct < 8; ++ct) {
#pragma unroll
        for (int r = 0; r < 8; ++r) o[ct][r] *= scl[r];
        v16h vf = frag_f16(&vTile[buf][ct * 16][0], 32, lane);
        o[ct] = wmma16(pf, vf, o[ct]);
      }
    }
  }

  // ---- normalize and store fp32 ----
  float* oB = out + (size_t)b * S * H;
#pragma unroll
  for (int ct = 0; ct < 8; ++ct)
#pragma unroll
    for (int r = 0; r < 8; ++r) {
      const int m = r + 8 * hf;
      oB[(size_t)(q0 + m) * H + ct * 16 + n] = o[ct][r] / lrow[r];
    }
}

extern "C" void kernel_launch(void* const* d_in, const int* in_sizes, int n_in,
                              void* d_out, int out_size, void* d_ws, size_t ws_size,
                              hipStream_t stream) {
  const float* q  = (const float*)d_in[0];
  const float* k  = (const float*)d_in[1];
  const float* v  = (const float*)d_in[2];
  const float* Wq = (const float*)d_in[3];
  const float* Wk = (const float*)d_in[4];
  const float* Wv = (const float*)d_in[5];
  float* out = (float*)d_out;

  // workspace layout (f16): qh 4MiB | kh 4MiB | vhT 4MiB | WqT/WkT/WvT 0.5MiB ea
  char* ws = (char*)d_ws;
  _Float16* qhW  = (_Float16*)(ws);
  _Float16* khW  = (_Float16*)(ws + (size_t)4 * 1024 * 1024);
  _Float16* vhTW = (_Float16*)(ws + (size_t)8 * 1024 * 1024);
  _Float16* WqT  = (_Float16*)(ws + (size_t)12 * 1024 * 1024);
  _Float16* WkT  = (_Float16*)(ws + (size_t)12 * 1024 * 1024 + 512 * 1024);
  _Float16* WvT  = (_Float16*)(ws + (size_t)13 * 1024 * 1024);

  {
    dim3 g((E * H + 255) / 256);
    wtrans_kernel<<<g, 256, 0, stream>>>(Wq, WqT);
    wtrans_kernel<<<g, 256, 0, stream>>>(Wk, WkT);
    wtrans_kernel<<<g, 256, 0, stream>>>(Wv, WvT);
  }
  {
    dim3 g(B * S / 64);  // 4 waves/WG, one 16-row strip each
    proj_kernel<<<g, 128, 0, stream>>>(q, WqT, qhW, 0);
    proj_kernel<<<g, 128, 0, stream>>>(k, WkT, khW, 0);
    proj_kernel<<<g, 128, 0, stream>>>(v, WvT, vhTW, 1);
  }
  {
    dim3 g(S / 64, B);
    attn_kernel<<<g, 128, 0, stream>>>(qhW, khW, vhTW, out);
  }
}
